// EndToEndHeteroGNN_35064113004690
// MI455X (gfx1250) — matrix operand order repaired and checked by reading, not compiled, not run on hardware
//
#include <hip/hip_runtime.h>
#include <hip/hip_bf16.h>

typedef __attribute__((ext_vector_type(16))) _Float16 v16h;
typedef __attribute__((ext_vector_type(8)))  _Float16 v8h;
typedef __attribute__((ext_vector_type(8)))  float    v8f;

union HF16 { v16h v; v8h h[2]; };

#define Bq    8
#define NAq   8192
#define NVq   2048
#define Hq    128
#define Kq    3
#define DEGq  16
#define NA_Tq (Bq*NAq)     // 65536
#define NV_Tq (Bq*NVq)     // 16384
#define E_AA  (NA_Tq*DEGq) // 1048576
#define E_VV  (NV_Tq*DEGq) // 262144
#define E_KNN (NV_Tq*Kq)   // 49152
#define LSTR  136          // padded LDS row stride in halfs (272B, 16B aligned, bank step 4)

// ---------------------------------------------------------------------------
// GEMM: Y[n,128] = X[n,128] @ W[128,128], f16 WMMA, f32 accumulate.
// Block = 128 threads (4 waves). Each block: 64 rows. Each wave: 16 rows.
// W staged TRANSPOSED in LDS so B fragments are contiguous 8-half runs.
// ---------------------------------------------------------------------------
__global__ __launch_bounds__(128)
void gemm128(const float* __restrict__ X, const float* __restrict__ W,
             float* __restrict__ Y, int nrows) {
  (void)nrows;
  __shared__ _Float16 Wt[128*LSTR];   // Wt[n*LSTR + k] = W[k*128 + n]
  __shared__ _Float16 Xl[64*LSTR];
  const int tid = threadIdx.x;
  const size_t r0 = (size_t)blockIdx.x * 64;
  for (int i = tid; i < 128*128; i += 128) {
    const int k = i >> 7, n = i & 127;
    Wt[n*LSTR + k] = (_Float16)W[i];
  }
  for (int i = tid; i < 64*128; i += 128) {
    const int r = i >> 7, c = i & 127;
    Xl[r*LSTR + c] = (_Float16)X[r0*128 + i];
  }
  __syncthreads();
  const int wave = tid >> 5, lane = tid & 31;
  const int l16 = lane & 15, hi = lane >> 4;
  // A fragments (16x32 f16 per k-step), ISA layout:
  // lane<16: K in {0..7, 16..23}; lane>=16: K in {8..15, 24..31}
  HF16 a[4];
  {
    const int ro = (wave*16 + l16) * LSTR;
    const int kb = hi ? 8 : 0;
#pragma unroll
    for (int ks = 0; ks < 4; ++ks) {
      a[ks].h[0] = *(const v8h*)&Xl[ro + ks*32 + kb];
      a[ks].h[1] = *(const v8h*)&Xl[ro + ks*32 + 16 + kb];
    }
  }
#pragma unroll
  for (int nt = 0; nt < 8; ++nt) {
    v8f acc = {};
    const int col = nt*16 + l16;
#pragma unroll
    for (int ks = 0; ks < 4; ++ks) {
      // B frag: lane<16: K in {0..7, 16..23}; lane>=16: K in {8..15, 24..31}
      const int kr = ks*32 + (hi ? 8 : 0);
      HF16 b;
      b.h[0] = *(const v8h*)&Wt[col*LSTR + kr];
      b.h[1] = *(const v8h*)&Wt[col*LSTR + kr + 16];
      acc = __builtin_amdgcn_wmma_f32_16x16x32_f16(false, a[ks].v, false, b.v,
                                                   (short)0, acc, false, false);
    }
    const size_t mrow = r0 + wave*16 + (hi ? 8 : 0);
#pragma unroll
    for (int r = 0; r < 8; ++r)
      Y[(mrow + r)*128 + col] = acc[r];
  }
}

// row self-norms: y[n] = dot(X[n,:128], X[n,:128]) : one wave per row
__global__ void rownorm(const float* __restrict__ X, float* __restrict__ y, int n) {
  const int wid  = (int)(((size_t)blockIdx.x*blockDim.x + threadIdx.x) >> 5);
  const int lane = threadIdx.x & 31;
  if (wid >= n) return;
  const float4 a = ((const float4*)(X + (size_t)wid*128))[lane];
  float s = a.x*a.x + a.y*a.y + a.z*a.z + a.w*a.w;
#pragma unroll
  for (int o = 16; o > 0; o >>= 1) s += __shfl_down(s, o, 32);
  if (lane == 0) y[wid] = s;
}

// ---------------------------------------------------------------------------
// kNN: per graph, per video node find K=3 nearest audio nodes.
// Rank by |a|^2 - 2*(v.a). Audio = A (M rows), video = B (N cols):
// each lane owns one video column -> per-lane top-3, shfl_xor(16) merge.
// Block = 128 threads (4 waves) handles 64 video nodes.
// Audio tiles double-buffered in LDS: one barrier per tile, global loads
// for tile t+1 overlap the 4 WMMAs of tile t. Audio norms precomputed.
// ---------------------------------------------------------------------------
__global__ __launch_bounds__(128)
void knn_topk(const float* __restrict__ XA, const float* __restrict__ XV,
              const float* __restrict__ anorm, int* __restrict__ knn_idx) {
  __shared__ _Float16 Vl[64*LSTR];
  __shared__ _Float16 Al[2][16*LSTR];
  const int tid  = threadIdx.x;
  const int b    = blockIdx.x >> 5;     // 32 vblocks per graph
  const int vblk = blockIdx.x & 31;
  const int v0   = vblk * 64;
  const float* xv = XV + ((size_t)b*NVq + v0) * 128;
  const float* xa = XA + (size_t)b*NAq*128;
  const float* an = anorm + (size_t)b*NAq;
  for (int i = tid; i < 64*128; i += 128) {
    const int r = i >> 7, c = i & 127;
    Vl[r*LSTR + c] = (_Float16)xv[i];
  }
  __syncthreads();
  const int wave = tid >> 5, lane = tid & 31;
  const int l16 = lane & 15, hi = lane >> 4;
  // Fixed B fragments: (K=dim, N=video), lane<16: K in {0..7, 16..23}; lane>=16: K in {8..15, 24..31}
  HF16 bf[4];
  {
    const int vo = (wave*16 + l16) * LSTR;
#pragma unroll
    for (int ks = 0; ks < 4; ++ks) {
      const int kr = ks*32 + (hi ? 8 : 0);
      bf[ks].h[0] = *(const v8h*)&Vl[vo + kr];
      bf[ks].h[1] = *(const v8h*)&Vl[vo + kr + 16];
    }
  }
  // preload tile 0 (each thread: 4 coalesced float4 = 16 elements)
  {
    float4 reg[4];
#pragma unroll
    for (int k = 0; k < 4; ++k) reg[k] = ((const float4*)xa)[tid + k*128];
#pragma unroll
    for (int k = 0; k < 4; ++k) {
      const int e = (tid + k*128) * 4;
      const int r = e >> 7, c = e & 127;
      _Float16* p = &Al[0][r*LSTR + c];
      p[0] = (_Float16)reg[k].x; p[1] = (_Float16)reg[k].y;
      p[2] = (_Float16)reg[k].z; p[3] = (_Float16)reg[k].w;
    }
  }
  __syncthreads();
  float best0 = 1e30f, best1 = 1e30f, best2 = 1e30f;
  int   idx0 = 0, idx1 = 0, idx2 = 0;
  const int kb = hi ? 8 : 0;
  const int NT = NAq/16;   // 512 tiles
  for (int at = 0; at < NT; ++at) {
    const int cur = at & 1;
    // issue next tile's global loads early (overlap with WMMA below)
    float4 reg[4];
    if (at + 1 < NT) {
      const float4* g = (const float4*)(xa + (size_t)(at+1)*16*128);
#pragma unroll
      for (int k = 0; k < 4; ++k) reg[k] = g[tid + k*128];
    }
    // per-lane audio norms for this tile (two cached float4 loads)
    const float4* an4 = (const float4*)(an + at*16 + (hi ? 8 : 0));
    const float4 n0 = an4[0], n1 = an4[1];
    const float anr[8] = {n0.x, n0.y, n0.z, n0.w, n1.x, n1.y, n1.z, n1.w};
    // 4 chained WMMAs on current buffer
    v8f acc = {};
#pragma unroll
    for (int ks = 0; ks < 4; ++ks) {
      HF16 af;
      af.h[0] = *(const v8h*)&Al[cur][l16*LSTR + ks*32 + kb];
      af.h[1] = *(const v8h*)&Al[cur][l16*LSTR + ks*32 + 16 + kb];
      acc = __builtin_amdgcn_wmma_f32_16x16x32_f16(false, af.v, false, bf[ks].v,
                                                   (short)0, acc, false, false);
    }
#pragma unroll
    for (int r = 0; r < 8; ++r) {
      const float sc = anr[r] - 2.0f*acc[r];
      const int id = b*NAq + at*16 + r + (hi ? 8 : 0);   // global audio index
      if (sc < best2) {
        if (sc < best0)      { best2=best1; idx2=idx1; best1=best0; idx1=idx0; best0=sc; idx0=id; }
        else if (sc < best1) { best2=best1; idx2=idx1; best1=sc; idx1=id; }
        else                 { best2=sc; idx2=id; }
      }
    }
    // convert + store next tile into alternate buffer
    if (at + 1 < NT) {
#pragma unroll
      for (int k = 0; k < 4; ++k) {
        const int e = (tid + k*128) * 4;
        const int r = e >> 7, c = e & 127;
        _Float16* p = &Al[1-cur][r*LSTR + c];
        p[0] = (_Float16)reg[k].x; p[1] = (_Float16)reg[k].y;
        p[2] = (_Float16)reg[k].z; p[3] = (_Float16)reg[k].w;
      }
    }
    __syncthreads();
  }
  // merge lane L with lane L+16 (same video column)
  float os0 = __shfl_xor(best0, 16, 32), os1 = __shfl_xor(best1, 16, 32), os2 = __shfl_xor(best2, 16, 32);
  int   oi0 = __shfl_xor(idx0, 16, 32),  oi1 = __shfl_xor(idx1, 16, 32),  oi2 = __shfl_xor(idx2, 16, 32);
  float cs[3] = {os0, os1, os2}; int ci[3] = {oi0, oi1, oi2};
#pragma unroll
  for (int k = 0; k < 3; ++k) {
    const float sc = cs[k]; const int id = ci[k];
    if (sc < best2) {
      if (sc < best0)      { best2=best1; idx2=idx1; best1=best0; idx1=idx0; best0=sc; idx0=id; }
      else if (sc < best1) { best2=best1; idx2=idx1; best1=sc; idx1=id; }
      else                 { best2=sc; idx2=id; }
    }
  }
  if (hi == 0) {
    const size_t v = (size_t)b*NVq + v0 + wave*16 + l16;
    knn_idx[v*3 + 0] = idx0;
    knn_idx[v*3 + 1] = idx1;
    knn_idx[v*3 + 2] = idx2;
  }
}

// ---------------------------------------------------------------------------
// Elementwise / aggregation helpers
// ---------------------------------------------------------------------------
__global__ void init_bias(float* __restrict__ Y, const float* __restrict__ b, size_t total) {
  const size_t stride = (size_t)gridDim.x * blockDim.x;
  for (size_t i = (size_t)blockIdx.x*blockDim.x + threadIdx.x; i < total; i += stride)
    Y[i] = b[i & 127];
}

__global__ void zero_f(float* __restrict__ p, size_t n) {
  const size_t stride = (size_t)gridDim.x * blockDim.x;
  for (size_t i = (size_t)blockIdx.x*blockDim.x + threadIdx.x; i < n; i += stride) p[i] = 0.f;
}

// out[dst] += H[src], one wave per edge, 4 channels per lane
__global__ void scatter_add(const float* __restrict__ Hm, const int* __restrict__ edges,
                            int E, float* __restrict__ Y) {
  const int wid  = (int)(((size_t)blockIdx.x*blockDim.x + threadIdx.x) >> 5);
  const int lane = threadIdx.x & 31;
  if (wid >= E) return;
  const int src = edges[wid];
  const int dst = edges[E + wid];
  const float* hs = Hm + (size_t)src*128;
  float* y = Y + (size_t)dst*128;
#pragma unroll
  for (int c = lane; c < 128; c += 32) atomicAdd(&y[c], hs[c]);
}

// per-graph sum / sumsq of relu(x); grid = B * blocksPerGraph
__global__ void stats_relu(const float* __restrict__ X, float* __restrict__ stats,
                           size_t per_graph, int blocksPerGraph) {
  const int b   = blockIdx.x / blocksPerGraph;
  const int blk = blockIdx.x % blocksPerGraph;
  const float* x = X + (size_t)b*per_graph;
  float s = 0.f, s2 = 0.f;
  const size_t stride = (size_t)blocksPerGraph * blockDim.x;
  for (size_t i = (size_t)blk*blockDim.x + threadIdx.x; i < per_graph; i += stride) {
    float v = x[i]; v = v > 0.f ? v : 0.f; s += v; s2 += v*v;
  }
  __shared__ float rs[256], rq[256];
  rs[threadIdx.x] = s; rq[threadIdx.x] = s2; __syncthreads();
  for (int o = 128; o > 0; o >>= 1) {
    if ((int)threadIdx.x < o) { rs[threadIdx.x] += rs[threadIdx.x+o]; rq[threadIdx.x] += rq[threadIdx.x+o]; }
    __syncthreads();
  }
  if (threadIdx.x == 0) { atomicAdd(&stats[b*2], rs[0]); atomicAdd(&stats[b*2+1], rq[0]); }
}

// Y = (relu(X)-mu)*rsqrt(var+eps)*g + be + resid
__global__ void apply_gln(const float* __restrict__ X, const float* __restrict__ resid,
                          const float* __restrict__ g, const float* __restrict__ be,
                          const float* __restrict__ stats, float* __restrict__ Y,
                          size_t per_graph, size_t total) {
  const size_t stride = (size_t)gridDim.x * blockDim.x;
  const float inv = 1.0f / (float)per_graph;
  for (size_t i = (size_t)blockIdx.x*blockDim.x + threadIdx.x; i < total; i += stride) {
    const size_t b = i / per_graph;
    const float mu  = stats[b*2] * inv;
    const float var = stats[b*2+1] * inv - mu*mu;
    const float rsq = rsqrtf(var + 1e-5f);
    float v = X[i]; v = fmaxf(v, 0.f);
    const int c = (int)(i & 127);
    Y[i] = (v - mu)*rsq*g[c] + be[c] + resid[i];
  }
}

// y[n] = dot(X[n,:128], w) : one wave per row
__global__ void rowdot(const float* __restrict__ X, const float* __restrict__ w,
                       float* __restrict__ y, int n) {
  const int wid  = (int)(((size_t)blockIdx.x*blockDim.x + threadIdx.x) >> 5);
  const int lane = threadIdx.x & 31;
  if (wid >= n) return;
  const float4 a = ((const float4*)(X + (size_t)wid*128))[lane];
  const float4 b = ((const float4*)w)[lane];
  float s = a.x*b.x + a.y*b.y + a.z*b.z + a.w*b.w;
#pragma unroll
  for (int o = 16; o > 0; o >>= 1) s += __shfl_down(s, o, 32);
  if (lane == 0) y[wid] = s;
}

// ---- GAT softmax over incoming edges per audio dst --------------------------
__device__ __forceinline__ unsigned enc_f(float f) {
  unsigned i = __float_as_uint(f);
  return (i & 0x80000000u) ? ~i : (i | 0x80000000u);
}
__device__ __forceinline__ float dec_f(unsigned e) {
  return (e & 0x80000000u) ? __uint_as_float(e & 0x7fffffffu) : __uint_as_float(~e);
}

__global__ void gat_max(const float* __restrict__ es, const float* __restrict__ ed,
                        const int* __restrict__ knn, unsigned* __restrict__ menc, int E2) {
  const int e = blockIdx.x*blockDim.x + threadIdx.x;
  if (e >= E2) return;
  const int src = e / 3;
  const int dst = knn[e];
  const float x = es[src] + ed[dst];
  const float l = x > 0.f ? x : 0.2f*x;
  atomicMax(&menc[dst], enc_f(l));
}

__global__ void gat_msg(const float* __restrict__ es, const float* __restrict__ ed,
                        const int* __restrict__ knn, const unsigned* __restrict__ menc,
                        const float* __restrict__ hs, float* __restrict__ den,
                        float* __restrict__ gat, int E2) {
  const int wid  = (int)(((size_t)blockIdx.x*blockDim.x + threadIdx.x) >> 5);
  const int lane = threadIdx.x & 31;
  if (wid >= E2) return;
  const int src = wid / 3;
  const int dst = knn[wid];
  const float x = es[src] + ed[dst];
  const float l = x > 0.f ? x : 0.2f*x;
  const float ex = __expf(l - dec_f(menc[dst]));
  if (lane == 0) atomicAdd(&den[dst], ex);
  const float* h = hs + (size_t)src*128;
  float* g = gat + (size_t)dst*128;
#pragma unroll
  for (int c = lane; c < 128; c += 32) atomicAdd(&g[c], ex*h[c]);
}

// Y = relu(Y + gat/den + b_gat)   (in place on GCN aggregate)
__global__ void combine_gat(float* __restrict__ Y, const float* __restrict__ gat,
                            const float* __restrict__ den, const float* __restrict__ bg,
                            size_t total) {
  const size_t stride = (size_t)gridDim.x * blockDim.x;
  for (size_t i = (size_t)blockIdx.x*blockDim.x + threadIdx.x; i < total; i += stride) {
    const size_t n = i >> 7; const int c = (int)(i & 127);
    const float d = den[n];
    const float gv = d > 0.f ? gat[i]/d : 0.f;
    float t = Y[i] + gv + bg[c];
    Y[i] = t > 0.f ? t : 0.f;
  }
}

// Global-attention readout: one block per graph
__global__ __launch_bounds__(256)
void readout(const float* __restrict__ X, const float* __restrict__ s,
             int n_per, float* __restrict__ out, int off) {
  const int b = blockIdx.x;
  const float* sb = s + (size_t)b*n_per;
  const float* xb = X + (size_t)b*n_per*128;
  __shared__ float red[256];
  const int tid = threadIdx.x;
  float mx = -1e30f;
  for (int n = tid; n < n_per; n += 256) mx = fmaxf(mx, sb[n]);
  red[tid] = mx; __syncthreads();
  for (int o = 128; o > 0; o >>= 1) { if (tid < o) red[tid] = fmaxf(red[tid], red[tid+o]); __syncthreads(); }
  mx = red[0]; __syncthreads();
  float se = 0.f;
  for (int n = tid; n < n_per; n += 256) se += __expf(sb[n]-mx);
  red[tid] = se; __syncthreads();
  for (int o = 128; o > 0; o >>= 1) { if (tid < o) red[tid] += red[tid+o]; __syncthreads(); }
  se = red[0]; __syncthreads();
  const int c = tid & 127, p = tid >> 7;
  float acc = 0.f;
  for (int n = p; n < n_per; n += 2) acc += __expf(sb[n]-mx) * xb[(size_t)n*128 + c];
  red[tid] = acc; __syncthreads();
  if (tid < 128) out[(size_t)b*256 + off + c] = (red[tid] + red[tid+128]) / se;
}

// ---------------------------------------------------------------------------
extern "C" void kernel_launch(void* const* d_in, const int* in_sizes, int n_in,
                              void* d_out, int out_size, void* d_ws, size_t ws_size,
                              hipStream_t stream) {
  (void)in_sizes; (void)n_in; (void)out_size; (void)ws_size;
  const float* x_audio = (const float*)d_in[0];
  const float* x_video = (const float*)d_in[1];
  const int*   edge_aa = (const int*)d_in[2];
  const int*   edge_vv = (const int*)d_in[3];
  const float* W_a0 = (const float*)d_in[4];  const float* b_a0 = (const float*)d_in[5];
  const float* W_v0 = (const float*)d_in[6];  const float* b_v0 = (const float*)d_in[7];
  const float* g_a0 = (const float*)d_in[8];  const float* be_a0 = (const float*)d_in[9];
  const float* g_v0 = (const float*)d_in[10]; const float* be_v0 = (const float*)d_in[11];
  const float* W_s1 = (const float*)d_in[12]; const float* b_s1 = (const float*)d_in[13];
  const float* Wg_src = (const float*)d_in[14]; const float* Wg_dst = (const float*)d_in[15];
  const float* a_src = (const float*)d_in[16];  const float* a_dst = (const float*)d_in[17];
  const float* b_gat = (const float*)d_in[18];
  const float* g_a1 = (const float*)d_in[19]; const float* be_a1 = (const float*)d_in[20];
  const float* g_v1 = (const float*)d_in[21]; const float* be_v1 = (const float*)d_in[22];
  const float* w_att_a = (const float*)d_in[23]; const float* w_att_v = (const float*)d_in[24];
  float* out = (float*)d_out;

  const size_t A = (size_t)NA_Tq * Hq;   // 8,388,608
  const size_t V = (size_t)NV_Tq * Hq;   // 2,097,152

  float* ws = (float*)d_ws;
  size_t off = 0;
  auto alloc = [&](size_t n) { float* p = ws + off; off += (n + 63) & ~(size_t)63; return p; };
  float* bufA1 = alloc(A);   // gemm out audio / hd / GAT accum
  float* bufA2 = alloc(A);   // GCN aggregation audio
  float* xa    = alloc(A);   // layer0 audio out
  float* xa1   = alloc(A);   // layer1 gemm scratch then final audio
  float* bufV1 = alloc(V);   // gemm out video / hs
  float* bufV2 = alloc(V);   // GCN aggregation video
  float* xv    = alloc(V);
  float* xv1   = alloc(V);
  float* es    = alloc(NV_Tq);
  float* ed    = alloc(NA_Tq);
  float* den   = alloc(NA_Tq);
  float* s_a   = alloc(NA_Tq);
  float* s_v   = alloc(NV_Tq);
  float* anorm = alloc(NA_Tq);
  float* stats = alloc(64);
  unsigned* menc = (unsigned*)alloc(NA_Tq);
  int* knn       = (int*)alloc((size_t)E_KNN);

  // ---- Layer 0: audio -------------------------------------------------------
  gemm128<<<NA_Tq/64, 128, 0, stream>>>(x_audio, W_a0, bufA1, NA_Tq);
  init_bias<<<8192, 256, 0, stream>>>(bufA2, b_a0, A);
  scatter_add<<<E_AA/8, 256, 0, stream>>>(bufA1, edge_aa, E_AA, bufA2);
  zero_f<<<1, 32, 0, stream>>>(stats, 16);
  stats_relu<<<Bq*64, 256, 0, stream>>>(bufA2, stats, (size_t)NAq*Hq, 64);
  apply_gln<<<8192, 256, 0, stream>>>(bufA2, x_audio, g_a0, be_a0, stats, xa, (size_t)NAq*Hq, A);

  // ---- Layer 0: video -------------------------------------------------------
  gemm128<<<NV_Tq/64, 128, 0, stream>>>(x_video, W_v0, bufV1, NV_Tq);
  init_bias<<<2048, 256, 0, stream>>>(bufV2, b_v0, V);
  scatter_add<<<E_VV/8, 256, 0, stream>>>(bufV1, edge_vv, E_VV, bufV2);
  zero_f<<<1, 32, 0, stream>>>(stats, 16);
  stats_relu<<<Bq*64, 256, 0, stream>>>(bufV2, stats, (size_t)NVq*Hq, 64);
  apply_gln<<<2048, 256, 0, stream>>>(bufV2, x_video, g_v0, be_v0, stats, xv, (size_t)NVq*Hq, V);

  // ---- GAT projections + kNN graph -----------------------------------------
  gemm128<<<NV_Tq/64, 128, 0, stream>>>(xv, Wg_src, bufV1, NV_Tq);    // hs
  rowdot<<<NV_Tq/8, 256, 0, stream>>>(bufV1, a_src, es, NV_Tq);
  gemm128<<<NA_Tq/64, 128, 0, stream>>>(xa, Wg_dst, bufA1, NA_Tq);    // hd
  rowdot<<<NA_Tq/8, 256, 0, stream>>>(bufA1, a_dst, ed, NA_Tq);
  rownorm<<<NA_Tq/8, 256, 0, stream>>>(xa, anorm, NA_Tq);
  knn_topk<<<Bq*(NVq/64), 128, 0, stream>>>(xa, xv, anorm, knn);

  zero_f<<<64, 256, 0, stream>>>((float*)menc, NA_Tq);
  zero_f<<<64, 256, 0, stream>>>(den, NA_Tq);
  zero_f<<<8192, 256, 0, stream>>>(bufA1, A);                         // GAT accumulator
  gat_max<<<(E_KNN+255)/256, 256, 0, stream>>>(es, ed, knn, menc, E_KNN);
  gat_msg<<<E_KNN/8, 256, 0, stream>>>(es, ed, knn, menc, bufV1, den, bufA1, E_KNN);

  // ---- Layer 1: audio (shared GCN + GAT) ------------------------------------
  gemm128<<<NA_Tq/64, 128, 0, stream>>>(xa, W_s1, xa1, NA_Tq);        // h_s1a (scratch)
  init_bias<<<8192, 256, 0, stream>>>(bufA2, b_s1, A);
  scatter_add<<<E_AA/8, 256, 0, stream>>>(xa1, edge_aa, E_AA, bufA2);
  combine_gat<<<8192, 256, 0, stream>>>(bufA2, bufA1, den, b_gat, A);
  zero_f<<<1, 32, 0, stream>>>(stats, 16);
  stats_relu<<<Bq*64, 256, 0, stream>>>(bufA2, stats, (size_t)NAq*Hq, 64);
  apply_gln<<<8192, 256, 0, stream>>>(bufA2, xa, g_a1, be_a1, stats, xa1, (size_t)NAq*Hq, A);

  // ---- Layer 1: video (shared GCN) ------------------------------------------
  gemm128<<<NV_Tq/64, 128, 0, stream>>>(xv, W_s1, bufV1, NV_Tq);
  init_bias<<<2048, 256, 0, stream>>>(bufV2, b_s1, V);
  scatter_add<<<E_VV/8, 256, 0, stream>>>(bufV1, edge_vv, E_VV, bufV2);
  zero_f<<<1, 32, 0, stream>>>(stats, 16);
  stats_relu<<<Bq*64, 256, 0, stream>>>(bufV2, stats, (size_t)NVq*Hq, 64);
  apply_gln<<<2048, 256, 0, stream>>>(bufV2, xv, g_v1, be_v1, stats, xv1, (size_t)NVq*Hq, V);

  // ---- Readout --------------------------------------------------------------
  rowdot<<<NA_Tq/8, 256, 0, stream>>>(xa1, w_att_a, s_a, NA_Tq);
  rowdot<<<NV_Tq/8, 256, 0, stream>>>(xv1, w_att_v, s_v, NV_Tq);
  readout<<<Bq, 256, 0, stream>>>(xa1, s_a, NAq, out, 0);
  readout<<<Bq, 256, 0, stream>>>(xv1, s_v, NVq, out, 128);
}
